// RDGT_29678224016040
// MI455X (gfx1250) — matrix operand order repaired and checked by reading, not compiled, not run on hardware
//
#include <hip/hip_runtime.h>
#include <stdint.h>

// ---------------------------------------------------------------------------
// Graphormer forward for MI455X (gfx1250, wave32, WMMA).
// All GEMMs run through v_wmma_f32_16x16x32_bf16. Attention is flash-style
// (no O(S^2) score materialization); bias is computed on the fly.
// A-tiles are staged global->LDS with the CDNA5 async path
// (global_load_async_to_lds_b128 + s_wait_asynccnt), overlapping the VALU
// f32->bf16 conversion of the weight tile.
// Workspace layout (~70 MB): h f32, y/q/k/v/o bf16, ffn-mid bf16, sp coeffs.
// ---------------------------------------------------------------------------

#define Bn   8
#define NNn  511
#define Sn   512
#define Dn   768
#define Hn   12
#define DHn  64
#define FFNn 3072
#define Ln   6
#define Mrows (Bn * Sn)   // 4096

typedef __attribute__((ext_vector_type(16))) __bf16 v16bf;
typedef __attribute__((ext_vector_type(8)))  float  v8f;

union Frag16 { uint4 u[2]; v16bf v; };      // 16 bf16 = 32B = 2 x uint4
union FragC  { v8f v; float f[8]; };        // 16x16 f32 C/D fragment

__device__ __forceinline__ unsigned short f2bf(float x) {
    union { float f; unsigned u; } cv; cv.f = x;
    unsigned u = cv.u;
    u = u + 0x7FFFu + ((u >> 16) & 1u);     // round-to-nearest-even
    return (unsigned short)(u >> 16);
}

// CDNA5 async global->LDS copy, 16 bytes per lane (ISA 08_async_tensor §4).
// dst_lds: LDS byte offset (low 32 bits of generic pointer, aperture rule);
// src: 64-bit global address. Tracked by ASYNCcnt.
__device__ __forceinline__ void async_cp_b128(unsigned dst_lds, const void* src) {
    asm volatile("global_load_async_to_lds_b128 %0, %1, off"
                 :: "v"(dst_lds), "v"((unsigned long long)(uintptr_t)src)
                 : "memory");
}
__device__ __forceinline__ void wait_async0() {
    asm volatile("s_wait_asynccnt 0x0" ::: "memory");
}

// ---------------------------------------------------------------------------
// Embedding: h[b,0,:] = graph_token ; h[b,s,:] = node_emb[x[b,s-1]]
// ---------------------------------------------------------------------------
__global__ void embed_kernel(const int* __restrict__ x,
                             const float* __restrict__ node_emb,
                             const float* __restrict__ gtok,
                             float* __restrict__ h) {
    int row = blockIdx.x;
    int b = row >> 9, s = row & (Sn - 1);
    float* out = h + (size_t)row * Dn;
    if (s == 0) {
        for (int d = threadIdx.x; d < Dn; d += blockDim.x) out[d] = gtok[d];
    } else {
        int idx = x[b * NNn + (s - 1)];
        const float* e = node_emb + (size_t)idx * Dn;
        for (int d = threadIdx.x; d < Dn; d += blockDim.x) out[d] = e[d];
    }
}

// ---------------------------------------------------------------------------
// Spatial-pos MLP collapses (1 input channel) to: bias = v*spA[h] + spC[h]
// ---------------------------------------------------------------------------
__global__ void spcoef_kernel(const float* __restrict__ w1, const float* __restrict__ b1,
                              const float* __restrict__ w2, const float* __restrict__ b2,
                              float* __restrict__ spAC) {
    int t = threadIdx.x;
    if (t < Hn) {
        float A = 0.f, C = 0.f;
        for (int c = 0; c < 128; ++c) {
            A += w1[c] * w2[c * Hn + t];
            C += b1[c] * w2[c * Hn + t];
        }
        spAC[t]      = A;
        spAC[16 + t] = C + b2[t];
    }
}

// ---------------------------------------------------------------------------
// LayerNorm over D=768 per row. outb!=null -> bf16 out, else f32 out.
// ---------------------------------------------------------------------------
__global__ void ln_kernel(const float* __restrict__ in,
                          const float* __restrict__ sc, const float* __restrict__ bi,
                          unsigned short* __restrict__ outb, float* __restrict__ outf) {
    __shared__ float r1[256];
    __shared__ float r2[256];
    int row = blockIdx.x, t = threadIdx.x;
    const float* p = in + (size_t)row * Dn;
    float v0 = p[t], v1 = p[t + 256], v2 = p[t + 512];
    r1[t] = v0 + v1 + v2;
    r2[t] = v0 * v0 + v1 * v1 + v2 * v2;
    __syncthreads();
    for (int o = 128; o > 0; o >>= 1) {
        if (t < o) { r1[t] += r1[t + o]; r2[t] += r2[t + o]; }
        __syncthreads();
    }
    float mean = r1[0] * (1.f / Dn);
    float var  = r2[0] * (1.f / Dn) - mean * mean;
    float inv  = rsqrtf(var + 1e-5f);
    float vv[3] = { v0, v1, v2 };
#pragma unroll
    for (int i = 0; i < 3; i++) {
        int d = t + i * 256;
        float y = (vv[i] - mean) * inv * sc[d] + bi[d];
        if (outb) outb[(size_t)row * Dn + d] = f2bf(y);
        else      outf[(size_t)row * Dn + d] = y;
    }
}

// ---------------------------------------------------------------------------
// Generic WMMA GEMM:  C[M][N] = A_bf16[M][K] * W_f32[K][N] + bias
// Block 64x64, BK=32, 128 threads = 4 waves, each wave 2x2 16x16 tiles.
// A tile staged via async global->LDS DMA, overlapped with W f32->bf16 VALU.
// mode 0: bf16 row-major out          mode 1: GELU(exact) -> bf16 row-major
// mode 2: *scale -> q/k layout [B,H,S,DH]   mode 3: v layout [B,H,DH,S]
// mode 4: f32 residual: resF[m][n] += val
// ---------------------------------------------------------------------------
__global__ __launch_bounds__(128)
void gemm_kernel(const unsigned short* __restrict__ A, const float* __restrict__ W,
                 const float* __restrict__ bias,
                 int K, int N, int mode, float scale,
                 unsigned short* __restrict__ outB, float* __restrict__ resF) {
    __shared__ __align__(16) unsigned short lA[64 * 40]; // [row][k], stride 40 (80B, 16B-mult)
    __shared__ __align__(16) unsigned short lB[64 * 40]; // [n][k]  transposed W tile

    int t = threadIdx.x;
    int mBase = blockIdx.y * 64, nBase = blockIdx.x * 64;
    int wave = t >> 5, lane = t & 31, half = lane >> 4, l15 = lane & 15;
    int wm = (wave >> 1) * 32, wn = (wave & 1) * 32;

    FragC c[2][2];
#pragma unroll
    for (int i = 0; i < 2; i++)
#pragma unroll
        for (int j = 0; j < 2; j++) {
            float bv = bias[nBase + wn + j * 16 + l15];
#pragma unroll
            for (int e = 0; e < 8; e++) c[i][j].f[e] = bv;
        }

    int arow = t >> 1, ac0 = (t & 1) * 16;     // A tile: 64 rows x 32, 16 elems/thread
    int bn = t >> 1, bkh = (t & 1) * 16;       // B tile: 64 n x 32 k, 16 elems/thread
    // LDS byte offsets for this thread's A-tile destination (2 x 16B chunks)
    unsigned adst0 = (unsigned)(uintptr_t)(void*)(lA + arow * 40 + ac0);
    unsigned adst1 = adst0 + 16u;

    for (int k0 = 0; k0 < K; k0 += 32) {
        // 1) kick off async DMA of the (already-bf16) A tile into LDS
        const unsigned short* asrc = A + (size_t)(mBase + arow) * K + k0 + ac0;
        async_cp_b128(adst0, asrc);
        async_cp_b128(adst1, asrc + 8);
        // 2) overlap: stage W with f32 -> bf16 conversion, transposed to [n][k]
#pragma unroll
        for (int kk = 0; kk < 16; ++kk) {
            float w = W[(size_t)(k0 + bkh + kk) * N + nBase + bn];
            lB[bn * 40 + bkh + kk] = f2bf(w);
        }
        // 3) drain DMA, then block barrier
        wait_async0();
        __syncthreads();

        Frag16 a[2], bfr[2];
#pragma unroll
        for (int i = 0; i < 2; i++) {           // A frag: ISA layout, two 16B chunks
            int r = wm + i * 16 + l15;
            a[i].u[0] = *(const uint4*)(lA + r * 40 + half * 8);
            a[i].u[1] = *(const uint4*)(lA + r * 40 + 16 + half * 8);
        }
#pragma unroll
        for (int j = 0; j < 2; j++) {           // B frag: ISA layout, two 16B chunks
            int n = wn + j * 16 + l15;
            const uint4* pb = (const uint4*)(lB + n * 40 + half * 8);
            bfr[j].u[0] = pb[0]; bfr[j].u[1] = pb[2];
        }
#pragma unroll
        for (int i = 0; i < 2; i++)
#pragma unroll
            for (int j = 0; j < 2; j++)
                c[i][j].v = __builtin_amdgcn_wmma_f32_16x16x32_bf16(
                    false, a[i].v, false, bfr[j].v, (short)0, c[i][j].v, false, false);
        __syncthreads();
    }

#pragma unroll
    for (int i = 0; i < 2; i++)
#pragma unroll
        for (int j = 0; j < 2; j++)
#pragma unroll
            for (int r = 0; r < 8; r++) {
                int mg = mBase + wm + i * 16 + r + half * 8;
                int ng = nBase + wn + j * 16 + l15;
                float v = c[i][j].f[r];
                if (mode == 0) {
                    outB[(size_t)mg * N + ng] = f2bf(v);
                } else if (mode == 1) {
                    float g = 0.5f * v * (1.0f + erff(v * 0.70710678118654752f));
                    outB[(size_t)mg * N + ng] = f2bf(g);
                } else if (mode == 2) {
                    v *= scale;
                    int b = mg >> 9, s = mg & (Sn - 1);
                    int hh = ng >> 6, dh = ng & 63;
                    outB[(((size_t)(b * Hn + hh)) * Sn + s) * DHn + dh] = f2bf(v);
                } else if (mode == 3) {
                    int b = mg >> 9, s = mg & (Sn - 1);
                    int hh = ng >> 6, dh = ng & 63;
                    outB[(((size_t)(b * Hn + hh)) * DHn + dh) * Sn + s] = f2bf(v);
                } else {
                    float* rp = resF + (size_t)mg * N + ng;
                    *rp = *rp + v;
                }
            }
}

// ---------------------------------------------------------------------------
// Flash attention, one wave = 16 query rows, streaming 32-key chunks.
// q: [B,H,S,DH] bf16 (pre-scaled), k: [B,H,S,DH] bf16, v: [B,H,DH,S] bf16.
// Bias: 2*attn_bias + (sp*spA[h]+spC[h]) on node block + virt_dist edges.
// ---------------------------------------------------------------------------
__global__ __launch_bounds__(128)
void attn_kernel(const unsigned short* __restrict__ qb, const unsigned short* __restrict__ kb,
                 const unsigned short* __restrict__ vb,
                 const float* __restrict__ attn_bias, const float* __restrict__ sp,
                 const float* __restrict__ spAC, const float* __restrict__ virt,
                 unsigned short* __restrict__ ob) {
    __shared__ __align__(16) unsigned short lP[4 * 16 * 32];  // per-wave P tile
    int b = blockIdx.z, h = blockIdx.y;
    int t = threadIdx.x;
    int wave = t >> 5, lane = t & 31, half = lane >> 4, l15 = lane & 15;
    int q0 = blockIdx.x * 64 + wave * 16;
    size_t bh = (size_t)(b * Hn + h);
    const unsigned short* qp = qb + bh * Sn * DHn;
    const unsigned short* kp = kb + bh * Sn * DHn;
    const unsigned short* vp = vb + bh * DHn * Sn;
    float sA = spAC[h], sC = spAC[16 + h], tb = virt[h];

    // Q fragments (resident for whole key loop): dh split 0..31 / 32..63
    Frag16 aq[2];
    {
        const unsigned short* qr = qp + (size_t)(q0 + l15) * DHn;
#pragma unroll
        for (int d = 0; d < 2; d++) {
            aq[d].u[0] = *(const uint4*)(qr + d * 32 + half * 8);
            aq[d].u[1] = *(const uint4*)(qr + d * 32 + 16 + half * 8);
        }
    }
    float m[8], l[8];
    FragC o[4];
#pragma unroll
    for (int r = 0; r < 8; r++) { m[r] = -3.0e38f; l[r] = 0.f; }
#pragma unroll
    for (int d = 0; d < 4; d++)
#pragma unroll
        for (int e = 0; e < 8; e++) o[d].f[e] = 0.f;

    unsigned short* P = lP + wave * 512;

    for (int kc = 0; kc < Sn; kc += 32) {
        FragC sc[2];
#pragma unroll
        for (int j = 0; j < 2; j++) {
#pragma unroll
            for (int e = 0; e < 8; e++) sc[j].f[e] = 0.f;
            // K^T B-frags read straight from k's row-major layout
            const unsigned short* kr = kp + (size_t)(kc + j * 16 + l15) * DHn;
            Frag16 bf0, bf1;
            bf0.u[0] = *(const uint4*)(kr + half * 8);
            bf0.u[1] = *(const uint4*)(kr + 16 + half * 8);
            bf1.u[0] = *(const uint4*)(kr + 32 + half * 8);
            bf1.u[1] = *(const uint4*)(kr + 32 + 16 + half * 8);
            sc[j].v = __builtin_amdgcn_wmma_f32_16x16x32_bf16(
                false, aq[0].v, false, bf0.v, (short)0, sc[j].v, false, false);
            sc[j].v = __builtin_amdgcn_wmma_f32_16x16x32_bf16(
                false, aq[1].v, false, bf1.v, (short)0, sc[j].v, false, false);
        }
        // bias + online softmax (row stats register-resident per 16-lane group)
#pragma unroll
        for (int r = 0; r < 8; r++) {
            int i = q0 + r + half * 8;
#pragma unroll
            for (int j = 0; j < 2; j++) {
                int jc = kc + j * 16 + l15;
                float s = sc[j].f[r] + 2.f * attn_bias[((size_t)b * Sn + i) * Sn + jc];
                if (i > 0 && jc > 0)
                    s += sp[((size_t)b * NNn + (i - 1)) * NNn + (jc - 1)] * sA + sC;
                if (i == 0 || jc == 0) s += tb;   // virt_dist edges (row0 OR col0,row>0)
                sc[j].f[r] = s;
            }
            float tm = fmaxf(sc[0].f[r], sc[1].f[r]);
#pragma unroll
            for (int off = 1; off < 16; off <<= 1) tm = fmaxf(tm, __shfl_xor(tm, off, 32));
            float mn = fmaxf(m[r], tm);
            float al = __expf(m[r] - mn);
            float p0 = __expf(sc[0].f[r] - mn);
            float p1 = __expf(sc[1].f[r] - mn);
            sc[0].f[r] = p0; sc[1].f[r] = p1;
            float rs = p0 + p1;
#pragma unroll
            for (int off = 1; off < 16; off <<= 1) rs += __shfl_xor(rs, off, 32);
            l[r] = l[r] * al + rs;
            m[r] = mn;
#pragma unroll
            for (int d = 0; d < 4; d++) o[d].f[r] *= al;    // rescale O
        }
        // P (16x32 bf16) through LDS to transpose C-layout -> A-frag layout
#pragma unroll
        for (int r = 0; r < 8; r++) {
            int prow = r + half * 8;
            P[prow * 32 + l15]      = f2bf(sc[0].f[r]);
            P[prow * 32 + 16 + l15] = f2bf(sc[1].f[r]);
        }
        __syncthreads();
        Frag16 ap;
        ap.u[0] = *(const uint4*)(P + l15 * 32 + half * 8);
        ap.u[1] = *(const uint4*)(P + l15 * 32 + 16 + half * 8);
#pragma unroll
        for (int d = 0; d < 4; d++) {
            // V^T B-frags read straight from v's [dh][s] layout
            const unsigned short* vr = vp + (size_t)(d * 16 + l15) * Sn + kc + half * 8;
            Frag16 bv2;
            bv2.u[0] = *(const uint4*)(vr);
            bv2.u[1] = *(const uint4*)(vr + 16);
            o[d].v = __builtin_amdgcn_wmma_f32_16x16x32_bf16(
                false, ap.v, false, bv2.v, (short)0, o[d].v, false, false);
        }
        __syncthreads();
    }
    // finalize: divide by l, write bf16 [B,S,D] for the o-projection GEMM
#pragma unroll
    for (int r = 0; r < 8; r++) {
        float inv = 1.f / l[r];
        int i = q0 + r + half * 8;
        size_t orow = ((size_t)b * Sn + i) * Dn + h * DHn;
#pragma unroll
        for (int d = 0; d < 4; d++)
            ob[orow + d * 16 + l15] = f2bf(o[d].f[r] * inv);
    }
}

// ---------------------------------------------------------------------------
extern "C" void kernel_launch(void* const* d_in, const int* in_sizes, int n_in,
                              void* d_out, int out_size, void* d_ws, size_t ws_size,
                              hipStream_t stream) {
    (void)in_sizes; (void)n_in; (void)out_size; (void)ws_size;
    const float* attn_bias = (const float*)d_in[0];
    const float* sp        = (const float*)d_in[1];
    const int*   x         = (const int*)d_in[2];
    const float* node_emb  = (const float*)d_in[3];
    const float* gtok      = (const float*)d_in[4];
    const float* virt      = (const float*)d_in[5];
    const float* sp_w1     = (const float*)d_in[6];
    const float* sp_b1     = (const float*)d_in[7];
    const float* sp_w2     = (const float*)d_in[8];
    const float* sp_b2     = (const float*)d_in[9];
    const float* ln1_s = (const float*)d_in[10];
    const float* ln1_b = (const float*)d_in[11];
    const float* wq = (const float*)d_in[12];
    const float* bq = (const float*)d_in[13];
    const float* wk = (const float*)d_in[14];
    const float* bk = (const float*)d_in[15];
    const float* wv = (const float*)d_in[16];
    const float* bv = (const float*)d_in[17];
    const float* wo = (const float*)d_in[18];
    const float* bo = (const float*)d_in[19];
    const float* ln2_s = (const float*)d_in[20];
    const float* ln2_b = (const float*)d_in[21];
    const float* fw1 = (const float*)d_in[22];
    const float* fb1 = (const float*)d_in[23];
    const float* fw2 = (const float*)d_in[24];
    const float* fb2 = (const float*)d_in[25];
    const float* fln_s = (const float*)d_in[26];
    const float* fln_b = (const float*)d_in[27];

    char* ws = (char*)d_ws;
    size_t off = 0;
    auto alloc = [&](size_t bytes) {
        void* p = ws + off;
        off += (bytes + 255) & ~(size_t)255;
        return p;
    };
    float*          h    = (float*)alloc((size_t)Mrows * Dn * 4);
    unsigned short* y    = (unsigned short*)alloc((size_t)Mrows * Dn * 2);
    unsigned short* qbf  = (unsigned short*)alloc((size_t)Mrows * Dn * 2);
    unsigned short* kbf  = (unsigned short*)alloc((size_t)Mrows * Dn * 2);
    unsigned short* vbf  = (unsigned short*)alloc((size_t)Mrows * Dn * 2);
    unsigned short* obf  = (unsigned short*)alloc((size_t)Mrows * Dn * 2);
    unsigned short* mid  = (unsigned short*)alloc((size_t)Mrows * FFNn * 2);
    float*          spAC = (float*)alloc(32 * 4);

    embed_kernel<<<Mrows, 256, 0, stream>>>(x, node_emb, gtok, h);
    spcoef_kernel<<<1, 32, 0, stream>>>(sp_w1, sp_b1, sp_w2, sp_b2, spAC);

    dim3 blk(128);
    dim3 gD(Dn / 64, Mrows / 64);    // 12 x 64
    dim3 gF(FFNn / 64, Mrows / 64);  // 48 x 64
    dim3 gA(Sn / 64, Hn, Bn);        // 8 x 12 x 8
    const float qscale = 0.125f;     // DH^-0.5

    for (int li = 0; li < Ln; ++li) {
        ln_kernel<<<Mrows, 256, 0, stream>>>(h, ln1_s + li * Dn, ln1_b + li * Dn, y, nullptr);
        gemm_kernel<<<gD, blk, 0, stream>>>(y, wq + (size_t)li * Dn * Dn, bq + li * Dn,
                                            Dn, Dn, 2, qscale, qbf, nullptr);
        gemm_kernel<<<gD, blk, 0, stream>>>(y, wk + (size_t)li * Dn * Dn, bk + li * Dn,
                                            Dn, Dn, 2, 1.0f, kbf, nullptr);
        gemm_kernel<<<gD, blk, 0, stream>>>(y, wv + (size_t)li * Dn * Dn, bv + li * Dn,
                                            Dn, Dn, 3, 1.0f, vbf, nullptr);
        attn_kernel<<<gA, blk, 0, stream>>>(qbf, kbf, vbf, attn_bias, sp, spAC, virt, obf);
        gemm_kernel<<<gD, blk, 0, stream>>>(obf, wo + (size_t)li * Dn * Dn, bo + li * Dn,
                                            Dn, Dn, 4, 1.0f, nullptr, h);
        ln_kernel<<<Mrows, 256, 0, stream>>>(h, ln2_s + li * Dn, ln2_b + li * Dn, y, nullptr);
        gemm_kernel<<<gF, blk, 0, stream>>>(y, fw1 + (size_t)li * Dn * FFNn, fb1 + li * FFNn,
                                            Dn, FFNn, 1, 1.0f, mid, nullptr);
        gemm_kernel<<<gD, blk, 0, stream>>>(mid, fw2 + (size_t)li * FFNn * Dn, fb2 + li * Dn,
                                            FFNn, Dn, 4, 1.0f, nullptr, h);
    }
    ln_kernel<<<Mrows, 256, 0, stream>>>(h, fln_s, fln_b, nullptr, (float*)d_out);
}
